// TransformerBlock_50379966382285
// MI455X (gfx1250) — matrix operand order repaired and checked by reading, not compiled, not run on hardware
//
#include <hip/hip_runtime.h>
#include <hip/hip_bf16.h>
#include <math.h>
#include <stdint.h>

// ---------------- problem constants (from reference) ----------------
constexpr int B_   = 4;
constexpr int S_   = 1024;
constexpr int E_   = 1024;
constexpr int H_   = 16;
constexpr int DFF_ = 4096;
constexpr int HD_  = 64;       // E/H
constexpr int MROWS = B_ * S_; // 4096 flattened rows

typedef __bf16 bf16_t;
typedef __attribute__((ext_vector_type(16))) __bf16 v16bf;
typedef __attribute__((ext_vector_type(8)))  __bf16 v8bf;
typedef __attribute__((ext_vector_type(4)))  __bf16 v4bf;
typedef __attribute__((ext_vector_type(8)))  float  v8f;
typedef __attribute__((ext_vector_type(4)))  unsigned u32x4;
typedef __attribute__((ext_vector_type(8)))  unsigned u32x8;

union FragAB {
    v16bf  v;
    v8bf   h2[2];
    bf16_t e[16];
};

__device__ __forceinline__ v8f wmma_bf16(v16bf a, v16bf b, v8f c) {
    return __builtin_amdgcn_wmma_f32_16x16x32_bf16(
        false, a, false, b, (short)0, c, false, false);
}

__device__ __forceinline__ float gelu_f(float x) {
    const float c = 0.7978845608028654f; // sqrt(2/pi)
    float t = c * (x + 0.044715f * x * x * x);
    return 0.5f * x * (1.0f + tanhf(t));
}

// one 16-byte async global->LDS copy (tracked by ASYNCcnt)
__device__ __forceinline__ void async_copy_b128(const void* gptr, void* lptr) {
    unsigned lds_off = (unsigned)(uintptr_t)lptr;   // low 32 bits of generic LDS ptr = LDS offset
    uint64_t gaddr   = (uint64_t)(uintptr_t)gptr;
    asm volatile("global_load_async_to_lds_b128 %0, %1, off"
                 :: "v"(lds_off), "v"(gaddr) : "memory");
}
__device__ __forceinline__ void wait_asynccnt0() {
    asm volatile("s_wait_asynccnt 0" ::: "memory");
}

// Tensor Data Mover: 2D tile load (global -> LDS), D# groups 0/1 per CDNA5 ISA
// tile_d0 elements contiguous (dim0), tile_d1 rows with stride stride_d0 (elements, data_size=2B)
__device__ __forceinline__ void tdm_load_2d(void* lptr, const void* gptr,
                                            unsigned tensor_d0, unsigned tensor_d1,
                                            unsigned tile_d0, unsigned tile_d1,
                                            unsigned stride_d0) {
    uint64_t ga = (uint64_t)(uintptr_t)gptr;
    unsigned lds_off = (unsigned)(uintptr_t)lptr;
    u32x4 g0;
    g0[0] = 1u;                                        // count=1 (valid), user mode
    g0[1] = lds_off;                                   // lds_addr [63:32]
    g0[2] = (unsigned)(ga & 0xffffffffu);              // global_addr low
    g0[3] = (unsigned)((ga >> 32) & 0x01ffffffu)       // global_addr[56:32]
          | 0x80000000u;                               // type=2 ("image") in [127:126]
    u32x8 g1;
    g1[0] = 1u << 16;                                  // data_size=1 (2 bytes); wg_mask=0
    g1[1] = (tensor_d0 & 0xffffu) << 16;               // tensor_dim0[15:0] @ bits 63:48
    g1[2] = (tensor_d0 >> 16) | ((tensor_d1 & 0xffffu) << 16);
    g1[3] = (tensor_d1 >> 16) | (tile_d0 << 16);       // tile_dim0 @ [127:112]
    g1[4] = tile_d1 & 0xffffu;                         // tile_dim1 @ [143:128]; tile_dim2=0
    g1[5] = stride_d0;                                 // tensor_dim0_stride[31:0]
    g1[6] = 0;                                         // stride hi / dim1_stride = 0
    g1[7] = 0;
    asm volatile("tensor_load_to_lds %0, %1" :: "s"(g0), "s"(g1) : "memory");
}

// ---------------- fp32 -> bf16 cast ----------------
__global__ void cast_f32_bf16(const float* __restrict__ in, bf16_t* __restrict__ out, int n) {
    int i = blockIdx.x * blockDim.x + threadIdx.x;
    if (i < n) out[i] = (bf16_t)in[i];
}

// ---------------- LayerNorm (ddof=1) -> bf16 ----------------
__global__ __launch_bounds__(256)
void layernorm_to_bf16(const float* __restrict__ x, const float* __restrict__ g,
                       const float* __restrict__ b, bf16_t* __restrict__ out) {
    __shared__ float redS[8];
    __shared__ float redQ[8];
    const int row = blockIdx.x;
    const float* xr = x + (size_t)row * E_;

    float s = 0.f, q = 0.f;
    for (int i = threadIdx.x; i < E_; i += 256) {
        float v = xr[i];
        s += v;
        q += v * v;
    }
    for (int off = 16; off > 0; off >>= 1) {
        s += __shfl_down(s, off, 32);
        q += __shfl_down(q, off, 32);
    }
    const int wid = threadIdx.x >> 5;
    const int lid = threadIdx.x & 31;
    if (lid == 0) { redS[wid] = s; redQ[wid] = q; }
    __syncthreads();
    if (threadIdx.x == 0) {
        float ts = 0.f, tq = 0.f;
        for (int w = 0; w < 8; ++w) { ts += redS[w]; tq += redQ[w]; }
        redS[0] = ts; redQ[0] = tq;
    }
    __syncthreads();
    const float mean = redS[0] * (1.0f / E_);
    const float var  = (redQ[0] - (float)E_ * mean * mean) * (1.0f / (E_ - 1));
    const float inv  = 1.0f / (sqrtf(var) + 1e-8f);

    bf16_t* outr = out + (size_t)row * E_;
    for (int i = threadIdx.x; i < E_; i += 256) {
        outr[i] = (bf16_t)(g[i] * (xr[i] - mean) * inv + b[i]);
    }
}

// ---------------- WMMA GEMM: TDM-staged B tile + async-staged A tile ----------------
// out[m,n] = act( sum_k A[m,k]*W[n,k] + bias[n] ) (+res)
// block = 256 threads (8 waves); wave tile 16x64; block tile 128x64
template<bool GELU, bool RES, bool OUTF32, bool OUTBF16, bool VTRANS>
__global__ __launch_bounds__(256)
void gemm_wmma(const bf16_t* __restrict__ A, const bf16_t* __restrict__ W,
               const float* __restrict__ bias, const float* __restrict__ res,
               float* __restrict__ outf, bf16_t* __restrict__ outb,
               int M, int N, int K) {
    __shared__ bf16_t sB[64 * 32];    // B tile: 64 cols x 32 K (4 KB)  <- TDM
    __shared__ bf16_t sA[128 * 32];   // A tile: 128 rows x 32 K (8 KB) <- async copies

    const int tid  = threadIdx.x;
    const int lane = tid & 31;
    const int wave = tid >> 5;
    const int hl   = lane >> 4;
    const int idx  = lane & 15;

    const int mb = blockIdx.y * 128;
    const int m0 = mb + wave * 16;
    const int n0 = blockIdx.x * 64;
    (void)M;

    // A staging: 512 x 16B chunks; thread does 2 (row = c/4, chunk = c%4)
    const int c0row = (tid * 2) >> 2, c0chk = (tid * 2) & 3;
    const int c1row = (tid * 2 + 1) >> 2, c1chk = (tid * 2 + 1) & 3;
    const bf16_t* ga0 = A + (size_t)(mb + c0row) * K + c0chk * 8;
    const bf16_t* ga1 = A + (size_t)(mb + c1row) * K + c1chk * 8;
    bf16_t* la0 = &sA[c0row * 32 + c0chk * 8];
    bf16_t* la1 = &sA[c1row * 32 + c1chk * 8];

    const bf16_t* wtile = W + (size_t)n0 * K;   // B tile origin (row n0, k 0)

    v8f acc[4] = {};
    const bf16_t* apr = &sA[(wave * 16 + idx) * 32 + hl * 16];

    for (int kk = 0; kk < K; kk += 32) {
        // stage B tile with one TDM DMA (wave 0), A tile with async copies (all threads)
        if (wave == 0) {
            tdm_load_2d(sB, wtile + kk, (unsigned)K, (unsigned)N, 32u, 64u, (unsigned)K);
        }
        async_copy_b128(ga0 + kk, la0);
        async_copy_b128(ga1 + kk, la1);
        __builtin_prefetch(ga0 + kk + 512, 0, 0);   // global_prefetch_b8
        wait_asynccnt0();
        if (wave == 0) __builtin_amdgcn_s_wait_tensorcnt((short)0);
        __syncthreads();

        v16bf afrag;
        {
            FragAB f;
            f.h2[0] = *(const v8bf*)(apr);
            f.h2[1] = *(const v8bf*)(apr + 8);
            afrag = f.v;
        }
#pragma unroll
        for (int t = 0; t < 4; ++t) {
            v16bf bfrag = *(const v16bf*)(&sB[(t * 16 + idx) * 32 + hl * 16]);
            acc[t] = wmma_bf16(afrag, bfrag, acc[t]);
        }
        __syncthreads();   // all waves done reading before next stage overwrites
    }

#pragma unroll
    for (int t = 0; t < 4; ++t) {
        const int col = n0 + t * 16 + idx;
        const float bv = bias[col];
#pragma unroll
        for (int r = 0; r < 8; ++r) {
            const int row = m0 + r + 8 * hl;
            float v = acc[t][r] + bv;
            if (GELU) v = gelu_f(v);
            if (RES)  v += res[(size_t)row * N + col];
            if (OUTF32)  outf[(size_t)row * N + col] = v;
            if (OUTBF16) {
                if (VTRANS) {
                    const int s  = row & (S_ - 1);
                    const int bb = row >> 10;
                    const int hh = col >> 6;
                    const int d  = col & (HD_ - 1);
                    outb[(((size_t)bb * H_ + hh) * HD_ + d) * S_ + s] = (bf16_t)v;
                } else {
                    outb[(size_t)row * N + col] = (bf16_t)v;
                }
            }
        }
    }
}

// ---------------- Attention: one wave per (b, h, 16-query tile) ----------------
__global__ __launch_bounds__(32)
void attention_wmma(const bf16_t* __restrict__ qb, const bf16_t* __restrict__ kb,
                    const bf16_t* __restrict__ vT, const int* __restrict__ mask,
                    bf16_t* __restrict__ ob) {
    extern __shared__ char smem[];
    float*  sc = (float*)smem;                    // [16][S_] fp32 scores (64 KB)
    bf16_t* sp = (bf16_t*)(smem + 16 * S_ * 4);   // [16][S_] bf16 probs  (32 KB)

    const int lane = threadIdx.x;
    const int hl   = lane >> 4;
    const int idx  = lane & 15;
    const int q0   = blockIdx.x * 16;
    const int h    = blockIdx.y;
    const int b    = blockIdx.z;
    const size_t bS = (size_t)b * S_;

    const bf16_t* qrow = qb + (bS + q0 + idx) * E_ + h * HD_ + hl * 16;

    // ---- Pass 1: S = Q K^T / 8 with mask -> LDS ----
    for (int n0 = 0; n0 < S_; n0 += 16) {
        v8f c = {};
#pragma unroll
        for (int kd = 0; kd < HD_; kd += 32) {
            v16bf afrag;
            {
                FragAB f;
                f.h2[0] = *(const v8bf*)(qrow + kd);
                f.h2[1] = *(const v8bf*)(qrow + kd + 8);
                afrag = f.v;
            }
            const bf16_t* kp = kb + (bS + n0 + idx) * E_ + h * HD_ + kd + hl * 16;
            v16bf bfrag = *(const v16bf*)kp;
            c = wmma_bf16(afrag, bfrag, c);
        }
#pragma unroll
        for (int r = 0; r < 8; ++r) {
            const int row = r + 8 * hl;
            const int col = n0 + idx;
            const int mv = mask[(size_t)b * S_ * S_ + (size_t)(q0 + row) * S_ + col];
            sc[row * S_ + col] = (mv == 0) ? -0.01f : c[r] * 0.125f;
        }
    }
    __syncthreads();

    // ---- Pass 2: row softmax, all 32 lanes (row=idx, column half=hl) ----
    {
        const int row = idx;
        const int c0  = hl * (S_ / 2);
        const float* srow = sc + row * S_ + c0;
        float mx = -3.0e38f;
        for (int j = 0; j < S_ / 2; j += 4) {
            float4 v = *(const float4*)(srow + j);
            mx = fmaxf(mx, fmaxf(fmaxf(v.x, v.y), fmaxf(v.z, v.w)));
        }
        mx = fmaxf(mx, __shfl_xor(mx, 16, 32));
        float sum = 0.f;
        for (int j = 0; j < S_ / 2; j += 4) {
            float4 v = *(const float4*)(srow + j);
            sum += __expf(v.x - mx) + __expf(v.y - mx) + __expf(v.z - mx) + __expf(v.w - mx);
        }
        sum += __shfl_xor(sum, 16, 32);
        const float inv = 1.0f / sum;
        bf16_t* prow = sp + row * S_ + c0;
        for (int j = 0; j < S_ / 2; j += 4) {
            float4 v = *(const float4*)(srow + j);
            v4bf p;
            p[0] = (bf16_t)(__expf(v.x - mx) * inv);
            p[1] = (bf16_t)(__expf(v.y - mx) * inv);
            p[2] = (bf16_t)(__expf(v.z - mx) * inv);
            p[3] = (bf16_t)(__expf(v.w - mx) * inv);
            *(v4bf*)(prow + j) = p;
        }
    }
    __syncthreads();

    // ---- Pass 3: O = P V  (contiguous B-operand from vT [B,H,HD,S]) ----
    const bf16_t* vhead = vT + ((size_t)b * H_ + h) * HD_ * S_;
    v8f acc[4] = {};
    for (int kk = 0; kk < S_; kk += 32) {
        v16bf afrag;
        {
            FragAB f;
            const bf16_t* pr = sp + idx * S_ + kk + hl * 16;
            f.h2[0] = *(const v8bf*)(pr);
            f.h2[1] = *(const v8bf*)(pr + 8);
            afrag = f.v;
        }
#pragma unroll
        for (int t = 0; t < 4; ++t) {
            const bf16_t* vp = vhead + (size_t)(t * 16 + idx) * S_ + kk + hl * 16;
            v16bf bfrag = *(const v16bf*)vp;
            acc[t] = wmma_bf16(afrag, bfrag, acc[t]);
        }
    }
#pragma unroll
    for (int t = 0; t < 4; ++t) {
#pragma unroll
        for (int r = 0; r < 8; ++r) {
            const int row = q0 + r + 8 * hl;
            ob[(bS + row) * E_ + h * HD_ + t * 16 + idx] = (bf16_t)acc[t][r];
        }
    }
}

// ---------------- host orchestration ----------------
extern "C" void kernel_launch(void* const* d_in, const int* in_sizes, int n_in,
                              void* d_out, int out_size, void* d_ws, size_t ws_size,
                              hipStream_t stream) {
    (void)in_sizes; (void)n_in; (void)out_size; (void)ws_size;

    const float* x    = (const float*)d_in[0];
    const int*   mask = (const int*)  d_in[1];
    const float* Wq   = (const float*)d_in[2];
    const float* bq   = (const float*)d_in[3];
    const float* Wk   = (const float*)d_in[4];
    const float* bk   = (const float*)d_in[5];
    const float* Wv   = (const float*)d_in[6];
    const float* bv   = (const float*)d_in[7];
    const float* Wo   = (const float*)d_in[8];
    const float* bo   = (const float*)d_in[9];
    const float* W1   = (const float*)d_in[10];
    const float* b1   = (const float*)d_in[11];
    const float* W2   = (const float*)d_in[12];
    const float* b2   = (const float*)d_in[13];
    const float* g1   = (const float*)d_in[14];
    const float* be1  = (const float*)d_in[15];
    const float* g2   = (const float*)d_in[16];
    const float* be2  = (const float*)d_in[17];
    float* out = (float*)d_out;

    char* ws = (char*)d_ws;
    size_t off = 0;
    auto take = [&](size_t bytes) { char* p = ws + off; off += (bytes + 255) & ~size_t(255); return p; };
    bf16_t* nx   = (bf16_t*)take((size_t)MROWS * E_ * 2);
    bf16_t* wqb  = (bf16_t*)take((size_t)E_ * E_ * 2);
    bf16_t* wkb  = (bf16_t*)take((size_t)E_ * E_ * 2);
    bf16_t* wvb  = (bf16_t*)take((size_t)E_ * E_ * 2);
    bf16_t* wob  = (bf16_t*)take((size_t)E_ * E_ * 2);
    bf16_t* w1b  = (bf16_t*)take((size_t)DFF_ * E_ * 2);
    bf16_t* w2b  = (bf16_t*)take((size_t)E_ * DFF_ * 2);
    bf16_t* qbuf = (bf16_t*)take((size_t)MROWS * E_ * 2);
    bf16_t* kbuf = (bf16_t*)take((size_t)MROWS * E_ * 2);
    bf16_t* vtb  = (bf16_t*)take((size_t)MROWS * E_ * 2);   // V in [B,H,HD,S]
    bf16_t* obuf = (bf16_t*)take((size_t)MROWS * E_ * 2);
    float*  x2   = (float*) take((size_t)MROWS * E_ * 4);
    bf16_t* nx2  = (bf16_t*)take((size_t)MROWS * E_ * 2);
    bf16_t* h1   = (bf16_t*)take((size_t)MROWS * DFF_ * 2);

    auto castN = [&](const float* src, bf16_t* dst, int n) {
        cast_f32_bf16<<<(n + 255) / 256, 256, 0, stream>>>(src, dst, n);
    };
    castN(Wq, wqb, E_ * E_);
    castN(Wk, wkb, E_ * E_);
    castN(Wv, wvb, E_ * E_);
    castN(Wo, wob, E_ * E_);
    castN(W1, w1b, DFF_ * E_);
    castN(W2, w2b, E_ * DFF_);

    layernorm_to_bf16<<<MROWS, 256, 0, stream>>>(x, g1, be1, nx);

    dim3 gQKV(E_ / 64, MROWS / 128);
    gemm_wmma<false, false, false, true, false><<<gQKV, 256, 0, stream>>>(nx, wqb, bq, nullptr, nullptr, qbuf, MROWS, E_, E_);
    gemm_wmma<false, false, false, true, false><<<gQKV, 256, 0, stream>>>(nx, wkb, bk, nullptr, nullptr, kbuf, MROWS, E_, E_);
    gemm_wmma<false, false, false, true, true ><<<gQKV, 256, 0, stream>>>(nx, wvb, bv, nullptr, nullptr, vtb,  MROWS, E_, E_);

    dim3 gAtt(S_ / 16, H_, B_);
    attention_wmma<<<gAtt, 32, 16 * S_ * 4 + 16 * S_ * 2, stream>>>(qbuf, kbuf, vtb, mask, obuf);

    gemm_wmma<false, true, true, false, false><<<gQKV, 256, 0, stream>>>(obuf, wob, bo, x, x2, nullptr, MROWS, E_, E_);

    layernorm_to_bf16<<<MROWS, 256, 0, stream>>>(x2, g2, be2, nx2);

    dim3 gFF1(DFF_ / 64, MROWS / 128);
    gemm_wmma<true, false, false, true, false><<<gFF1, 256, 0, stream>>>(nx2, w1b, b1, nullptr, nullptr, h1, MROWS, DFF_, E_);

    dim3 gFF2(E_ / 64, MROWS / 128);
    gemm_wmma<false, true, true, false, false><<<gFF2, 256, 0, stream>>>(h1, w2b, b2, x2, out, nullptr, MROWS, E_, DFF_);
}
